// AdaptiveResonanceNetwork_81441169867033
// MI455X (gfx1250) — matrix operand order repaired and checked by reading, not compiled, not run on hardware
//
#include <hip/hip_runtime.h>
#include <math.h>

// ---------------------------------------------------------------------------
// AdaptiveResonanceNetwork fused pipeline for gfx1250 (MI455X, wave32, WMMA).
//
//  * bf16 WMMA (v_wmma_f32_16x16x32_bf16) for all GEMMs, f32 accumulate.
//  * Weights pre-transposed to bf16 W^T[n][k] in workspace (L2 resident);
//    B-fragments are global_load_b128 (addrspace(1) forced, no flat).
//  * All LDS row strides padded so (stride mod 64 dwords) avoids bank
//    conflicts: fragment reads (16 lanes, row-strided b128) and column
//    scans (32 lanes, row-strided b32) are conflict-free on 64 banks.
//  * Every GEMM decomposed into (N-tile x M-half) units: balanced over
//    8 waves, 4 accumulators per unit.
//  * Back end collapses to a 64-bin SOFM winner histogram + tiny finalize.
// ---------------------------------------------------------------------------

typedef __bf16 bf16_t;
typedef __bf16 v16bf __attribute__((ext_vector_type(16)));
typedef __bf16 v4bf  __attribute__((ext_vector_type(4)));
typedef float  v8f   __attribute__((ext_vector_type(8)));
typedef float  v4f   __attribute__((ext_vector_type(4)));
typedef unsigned int v4u __attribute__((ext_vector_type(4)));  // plain 128b carrier

__device__ __forceinline__ bf16_t tobf(float x) { return (bf16_t)x; }

// Force a generic pointer to the global address space.
#define GCAST(T, p) ((const __attribute__((address_space(1))) T*)(unsigned long long)(p))

union V16U { v16bf v; v4u q[2]; };

// 16-element bf16 fragment from LDS (two 16B chunks).
__device__ __forceinline__ v16bf ld_frag(const bf16_t* p0, const bf16_t* p1) {
  V16U t;
  t.q[0] = *(const v4u*)p0;
  t.q[1] = *(const v4u*)p1;
  return t.v;
}

// 16-element bf16 fragment from GLOBAL memory (global_load_b128 x2).
__device__ __forceinline__ v16bf ld_frag_g(const bf16_t* p0, const bf16_t* p1) {
  V16U t;
  t.q[0] = *GCAST(v4u, p0);
  t.q[1] = *GCAST(v4u, p1);
  return t.v;
}

__device__ __forceinline__ v4f ldg_f4(const float* p) {
  return *GCAST(v4f, p);
}

// ---- LDS strides (elements). All multiples of 8 (16B-aligned rows) and
// chosen so stride-in-dwords mod 64 has small gcd with 64 (bank spread).
#define LDA_FE 584   // FEATS bf16 [128][584]   (292 dw, mod64=36, gcd 4)
#define LDA_QP 264   // QPAD  bf16 [128][264]   (132 dw, mod64=4)
#define LDA_AP 136   // APAD  bf16 [128][136]   ( 68 dw, mod64=4)
#define LDA_OB 200   // OB/FUSEDB bf16 [128][200](100 dw, mod64=36)
#define LDF    193   // ACTF  f32  [128][193]   (mod64=1: conflict-free scans)
#define LDS_SC 68    // SCORES f32 [128][68]
#define LDS_DI 65    // DIST   f32 [128][65]

// Generic 128xK @ KxN GEMM. A in LDS (bf16 row-major, stride lda, col base
// acol); B^T in global bf16 [n][k]; f32 result to LDS OUT (stride ldo).
// Work unit u = (N-tile, M-half): ntiles*2 units striped over 8 waves.
__device__ __forceinline__ void gemm_nt(
    const bf16_t* __restrict__ A, int lda, int acol,
    const bf16_t* __restrict__ BT, int K,
    float* __restrict__ OUT, int ldo, int ntiles,
    int wave, int ln, int kh)
{
  const int nunits = ntiles * 2;
  for (int u = wave; u < nunits; u += 8) {
    const int nt = u >> 1;
    const int mh = (u & 1) * 4;                  // M-subtile base (0 or 4)
    v8f acc[4] = {};
    const bf16_t* bcol  = BT + (size_t)(nt * 16 + ln) * K + kh * 8;
    const bf16_t* abase = A + (size_t)(mh * 16 + ln) * lda + acol + kh * 8;
    for (int k0 = 0; k0 < K; k0 += 32) {
      v16bf bfr = ld_frag_g(bcol + k0, bcol + k0 + 16);
#pragma unroll
      for (int m = 0; m < 4; ++m) {
        const bf16_t* ap = abase + (size_t)(m * 16) * lda + k0;
        v16bf afr = ld_frag(ap, ap + 16);
        acc[m] = __builtin_amdgcn_wmma_f32_16x16x32_bf16(
            false, afr, false, bfr, (short)0, acc[m], false, false);
      }
    }
#pragma unroll
    for (int m = 0; m < 4; ++m)
#pragma unroll
      for (int j = 0; j < 8; ++j)
        OUT[(size_t)((mh + m) * 16 + j + 8 * kh) * ldo + nt * 16 + ln] = acc[m][j];
  }
}

// LayerNorm (+pre-bias) + exact GELU; X is ACTF (stride LDF) -> bf16 out.
__device__ __forceinline__ void ln_gelu_store(
    const float* __restrict__ X, const float* __restrict__ b,
    const float* __restrict__ g, const float* __restrict__ bb,
    bf16_t* __restrict__ OUTB, int ldo, int ocol,
    float* __restrict__ STAT, int tid)
{
  if (tid < 128) {
    float s = 0.f, s2 = 0.f;
    for (int c = 0; c < 192; ++c) {
      float v = X[tid * LDF + c] + b[c];
      s += v; s2 += v * v;
    }
    float m   = s  * (1.0f / 192.0f);
    float var = s2 * (1.0f / 192.0f) - m * m;
    STAT[tid * 2 + 0] = m;
    STAT[tid * 2 + 1] = rsqrtf(var + 1e-5f);
  }
  __syncthreads();
  for (int idx = tid; idx < 128 * 192; idx += 256) {
    int r = idx / 192, c = idx - r * 192;
    float v = X[r * LDF + c] + b[c];
    v = (v - STAT[r * 2]) * STAT[r * 2 + 1] * g[c] + bb[c];
    v = 0.5f * v * (1.0f + erff(v * 0.70710678118654752f));
    OUTB[(size_t)r * ldo + ocol + c] = tobf(v);
  }
}

// ---------------- prep kernels ----------------

__device__ __forceinline__ void tconv(const float* __restrict__ W,
                                      bf16_t* __restrict__ WT, int K,
                                      int tid, int nth) {
  int tot = K * 192;
  for (int idx = tid; idx < tot; idx += nth) {
    int k = idx / 192, n = idx - k * 192;
    WT[(size_t)n * K + k] = tobf(W[idx]);
  }
}

__global__ void arn_prep_weights(
    const float* __restrict__ ew0, const float* __restrict__ ew1,
    const float* __restrict__ ew2, const float* __restrict__ fw,
    const float* __restrict__ wq,  const float* __restrict__ wo,
    const float* __restrict__ sofm,
    bf16_t* __restrict__ encT0, bf16_t* __restrict__ encT1,
    bf16_t* __restrict__ encT2, bf16_t* __restrict__ fusT,
    bf16_t* __restrict__ wqT,   bf16_t* __restrict__ woT,
    bf16_t* __restrict__ gridB, float* __restrict__ gridnorm,
    unsigned* __restrict__ hist)
{
  int tid = blockIdx.x * blockDim.x + threadIdx.x;
  int nth = gridDim.x * blockDim.x;
  for (int g = tid; g < 64; g += nth) {
    hist[g] = 0u;
    float s = 0.f;
    for (int c = 0; c < 192; ++c) { float v = sofm[g * 192 + c]; s += v * v; }
    gridnorm[g] = s;
  }
  tconv(ew0, encT0, 64,  tid, nth);
  tconv(ew1, encT1, 256, tid, nth);
  tconv(ew2, encT2, 128, tid, nth);
  tconv(fw,  fusT,  576, tid, nth);
  for (int l = 0; l < 3; ++l) {
    tconv(wq + l * 36864, wqT + (size_t)l * 36864, 192, tid, nth);
    tconv(wo + l * 36864, woT + (size_t)l * 36864, 192, tid, nth);
  }
  for (int idx = tid; idx < 64 * 192; idx += nth) gridB[idx] = tobf(sofm[idx]);
}

// K/V of the attention: k = mem@wk + bk, v = mem@wv + bv, stored transposed &
// padded for WMMA B-fragments:
//   kT[i][h][slot(16)][kd(64)]  (kd>=48 zero)  -- B^T for scores GEMM (K=64)
//   vT[i][h][n(48)][kk(32)]     (kk>=16 zero)  -- B^T for attn@V GEMM (K=32)
__global__ void arn_prep_attn(
    const float* __restrict__ mem, const float* __restrict__ wk,
    const float* __restrict__ wv,  const float* __restrict__ bk,
    const float* __restrict__ bv,
    bf16_t* __restrict__ kT, bf16_t* __restrict__ vT)
{
  int tid = blockIdx.x * blockDim.x + threadIdx.x;
  int nth = gridDim.x * blockDim.x;
  for (int idx = tid; idx < 3 * 4 * 16 * 64; idx += nth) {
    int i = idx / 4096, r = idx % 4096;
    int h = r / 1024, r2 = r % 1024;
    int slot = r2 / 64, kd = r2 % 64;
    float val = 0.f;
    if (kd < 48) {
      int c = h * 48 + kd;
      float acc = bk[i * 192 + c];
      const float* mrow = mem + (size_t)(i * 16 + slot) * 192;
      const float* wcol = wk + (size_t)i * 36864 + c;
      for (int d = 0; d < 192; ++d) acc += mrow[d] * wcol[(size_t)d * 192];
      val = acc;
    }
    kT[idx] = tobf(val);
  }
  for (int idx = tid; idx < 3 * 4 * 48 * 32; idx += nth) {
    int i = idx / 6144, r = idx % 6144;
    int h = r / 1536, r2 = r % 1536;
    int n = r2 / 32, kk = r2 % 32;
    float val = 0.f;
    if (kk < 16) {
      int c = h * 48 + n;
      float acc = bv[i * 192 + c];
      const float* mrow = mem + (size_t)(i * 16 + kk) * 192;
      const float* wcol = wv + (size_t)i * 36864 + c;
      for (int d = 0; d < 192; ++d) acc += mrow[d] * wcol[(size_t)d * 192];
      val = acc;
    }
    vT[idx] = tobf(val);
  }
}

// ---------------- main fused pipeline ----------------
// LDS layout (bytes):
//   FEATS region @0        : 128*584*2 = 149504
//     QPAD   @0      (128*264*2 = 67584)
//     SCORES @67584  (128*68*4  = 34816)  -> ends 102400
//     APAD   @102400 (128*136*2 = 34816)  -> ends 137216
//     OB     @0      (128*200*2 = 51200)
//     DIST   @67584  (128*65*4  = 33280)
//   XBUF/FUSEDB @149504    : 128*264*2 = 67584 (max, K=256 padded)
//   ACTF @217088           : 128*193*4 = 98816
//   STAT @315904           : 1024
#define SM_FEATS 0
#define SM_XBUF  149504
#define SM_ACTF  217088
#define SM_STAT  315904
#define SMEM_BYTES 316928

__global__ __launch_bounds__(256, 1) void arn_fused_pipeline(
    const float* __restrict__ x_vib, const float* __restrict__ x_aco,
    const float* __restrict__ x_tmp,
    const float* __restrict__ eb0, const float* __restrict__ eg0, const float* __restrict__ ebb0,
    const float* __restrict__ eb1, const float* __restrict__ eg1, const float* __restrict__ ebb1,
    const float* __restrict__ eb2, const float* __restrict__ eg2, const float* __restrict__ ebb2,
    const float* __restrict__ fus_b, const float* __restrict__ fus_g, const float* __restrict__ fus_bb,
    const float* __restrict__ res_bq, const float* __restrict__ res_bo,
    const bf16_t* __restrict__ encT0, const bf16_t* __restrict__ encT1,
    const bf16_t* __restrict__ encT2, const bf16_t* __restrict__ fusT,
    const bf16_t* __restrict__ wqT,   const bf16_t* __restrict__ woT,
    const bf16_t* __restrict__ kT,    const bf16_t* __restrict__ vT,
    const bf16_t* __restrict__ gridB, const float* __restrict__ gridnorm,
    unsigned* __restrict__ hist)
{
  extern __shared__ char smem[];
  bf16_t* FEATS  = (bf16_t*)(smem + SM_FEATS);          // [128][584]
  bf16_t* XBUF   = (bf16_t*)(smem + SM_XBUF);           // [128][K+8]
  bf16_t* FUSEDB = XBUF;                                // [128][200]
  float*  ACTF   = (float*) (smem + SM_ACTF);           // [128][193]
  float*  STAT   = (float*) (smem + SM_STAT);           // [128][2]
  bf16_t* QPAD   = (bf16_t*)(smem + SM_FEATS);          // [128][264]
  float*  SCORES = (float*) (smem + SM_FEATS + 67584);  // [128][68]
  bf16_t* APAD   = (bf16_t*)(smem + SM_FEATS + 102400); // [128][136]
  bf16_t* OB     = (bf16_t*)(smem + SM_FEATS);          // [128][200]
  float*  DIST   = (float*) (smem + SM_FEATS + 67584);  // [128][65]

  const int tid  = threadIdx.x;
  const int wave = tid >> 5;
  const int lane = tid & 31;
  const int ln   = lane & 15;
  const int kh   = lane >> 4;
  const size_t row0 = (size_t)blockIdx.x * 128;
  const float ATT_SCALE = 0.14433756729740643f;   // 1/sqrt(48)

  const float*  xsrc[3] = {x_vib, x_aco, x_tmp};
  const int     Ks[3]   = {64, 256, 128};
  const bf16_t* ets[3]  = {encT0, encT1, encT2};
  const float*  ebs[3]  = {eb0, eb1, eb2};
  const float*  egs[3]  = {eg0, eg1, eg2};
  const float*  ebbs[3] = {ebb0, ebb1, ebb2};

  // ---- per-modality encoders: Linear -> LN -> GELU -> feats (bf16) ----
  for (int mod = 0; mod < 3; ++mod) {
    const int K = Ks[mod];
    const int ldx = K + 8;                         // padded LDS stride
    const float* xg = xsrc[mod] + row0 * K;
    const int qperrow = K >> 2;
    for (int idx = tid; idx < 128 * qperrow; idx += 256) {
      int r = idx / qperrow, c4 = (idx - r * qperrow) * 4;
      v4f f = ldg_f4(xg + (size_t)r * K + c4);
      v4bf b;
#pragma unroll
      for (int j = 0; j < 4; ++j) b[j] = tobf(f[j]);
      *(v4bf*)(XBUF + (size_t)r * ldx + c4) = b;
    }
    // warm next modality's rows into cache while this GEMM runs
    if (mod < 2) {
      const int Kn = Ks[mod + 1];
      const float* xn = xsrc[mod + 1] + row0 * Kn;
      for (int off = tid * 16; off < 128 * Kn; off += 256 * 16)
        __builtin_prefetch(xn + off, 0, 1);
    }
    __syncthreads();
    gemm_nt(XBUF, ldx, 0, ets[mod], K, ACTF, LDF, 12, wave, ln, kh);
    __syncthreads();
    ln_gelu_store(ACTF, ebs[mod], egs[mod], ebbs[mod],
                  FEATS, LDA_FE, mod * 192, STAT, tid);
    __syncthreads();
  }

  // ---- fusion: feats[128x576] @ fus_w -> LN -> GELU -> fusedB ----
  gemm_nt(FEATS, LDA_FE, 0, fusT, 576, ACTF, LDF, 12, wave, ln, kh);
  __syncthreads();
  ln_gelu_store(ACTF, fus_b, fus_g, fus_bb, FUSEDB, LDA_OB, 0, STAT, tid);
  __syncthreads();

  // ---- 3 adaptive resonance layers ----
  for (int L = 0; L < 3; ++L) {
    // q = fused @ wq
    gemm_nt(FUSEDB, LDA_OB, 0, wqT + (size_t)L * 36864, 192, ACTF, LDF, 12,
            wave, ln, kh);
    __syncthreads();
    // q + bq, pad head-dim 48 -> 64 (bf16)
    for (int idx = tid; idx < 128 * 256; idx += 256) {
      int r = idx >> 8, c = idx & 255;
      int h = c >> 6, kd = c & 63;
      float v = 0.f;
      if (kd < 48) {
        int cc = h * 48 + kd;
        v = ACTF[r * LDF + cc] + res_bq[L * 192 + cc];
      }
      QPAD[(size_t)r * LDA_QP + c] = tobf(v);
    }
    __syncthreads();
    // scores[b,h,slot] = q_h . k_h : 8 units = (head x M-half), one per wave
    {
      const int h  = wave >> 1;
      const int mh = (wave & 1) * 4;
      v8f acc[4] = {};
      const bf16_t* bcol  = kT + ((size_t)((L * 4 + h) * 16) + ln) * 64 + kh * 8;
      const bf16_t* abase = QPAD + (size_t)(mh * 16 + ln) * LDA_QP + h * 64 + kh * 8;
      for (int k0 = 0; k0 < 64; k0 += 32) {
        v16bf bfr = ld_frag_g(bcol + k0, bcol + k0 + 16);
#pragma unroll
        for (int m = 0; m < 4; ++m) {
          const bf16_t* ap = abase + (size_t)(m * 16) * LDA_QP + k0;
          v16bf afr = ld_frag(ap, ap + 16);
          acc[m] = __builtin_amdgcn_wmma_f32_16x16x32_bf16(
              false, afr, false, bfr, (short)0, acc[m], false, false);
        }
      }
#pragma unroll
      for (int m = 0; m < 4; ++m)
#pragma unroll
        for (int j = 0; j < 8; ++j)
          SCORES[(size_t)((mh + m) * 16 + j + 8 * kh) * LDS_SC + h * 16 + ln] = acc[m][j];
    }
    __syncthreads();
    // softmax over 16 slots (f32), write attn padded 16 -> 32 (bf16)
    for (int p = tid; p < 512; p += 256) {
      int r = p >> 2, h = p & 3;
      const float* sc = SCORES + r * LDS_SC + h * 16;
      float v[16];
      float mx = -3.4e38f;
#pragma unroll
      for (int s = 0; s < 16; ++s) { v[s] = sc[s] * ATT_SCALE; mx = fmaxf(mx, v[s]); }
      float sum = 0.f;
#pragma unroll
      for (int s = 0; s < 16; ++s) { v[s] = expf(v[s] - mx); sum += v[s]; }
      float inv = 1.0f / sum;
      bf16_t* ap = APAD + (size_t)r * LDA_AP + h * 32;
#pragma unroll
      for (int s = 0; s < 16; ++s) ap[s] = tobf(v[s] * inv);
#pragma unroll
      for (int s = 16; s < 32; ++s) ap[s] = tobf(0.f);
    }
    __syncthreads();
    // o = attn @ V : 24 units = (head x 16-col tile x M-half)
    for (int u = wave; u < 24; u += 8) {
      int h = u / 6, rem = u - h * 6;
      int t = rem >> 1, mh = (rem & 1) * 4;
      v8f acc[4] = {};
      const bf16_t* bcol = vT + ((size_t)((L * 4 + h) * 48) + t * 16 + ln) * 32 + kh * 8;
      v16bf bfr = ld_frag_g(bcol, bcol + 16);
      const bf16_t* abase = APAD + (size_t)(mh * 16 + ln) * LDA_AP + h * 32 + kh * 8;
#pragma unroll
      for (int m = 0; m < 4; ++m) {
        const bf16_t* ap = abase + (size_t)(m * 16) * LDA_AP;
        v16bf afr = ld_frag(ap, ap + 16);
        acc[m] = __builtin_amdgcn_wmma_f32_16x16x32_bf16(
            false, afr, false, bfr, (short)0, acc[m], false, false);
      }
#pragma unroll
      for (int m = 0; m < 4; ++m)
#pragma unroll
        for (int j = 0; j < 8; ++j)
          ACTF[(size_t)((mh + m) * 16 + j + 8 * kh) * LDF + h * 48 + t * 16 + ln] = acc[m][j];
    }
    __syncthreads();
    // o -> bf16 (QPAD region is dead now)
    for (int idx = tid; idx < 128 * 192; idx += 256) {
      int r = idx / 192, c = idx - r * 192;
      OB[(size_t)r * LDA_OB + c] = tobf(ACTF[r * LDF + c]);
    }
    __syncthreads();
    // fused_pre = o @ wo
    gemm_nt(OB, LDA_OB, 0, woT + (size_t)L * 36864, 192, ACTF, LDF, 12,
            wave, ln, kh);
    __syncthreads();
    // fused = fused_pre + bo (bf16 copy for next GEMM; f32 stays in ACTF)
    for (int idx = tid; idx < 128 * 192; idx += 256) {
      int r = idx / 192, c = idx - r * 192;
      FUSEDB[(size_t)r * LDA_OB + c] = tobf(ACTF[r * LDF + c] + res_bo[L * 192 + c]);
    }
    __syncthreads();
  }

  // ---- SOFM: dot = fused @ grid^T (K=192, N=64), then argmin + histogram ----
  gemm_nt(FUSEDB, LDA_OB, 0, gridB, 192, DIST, LDS_DI, 4, wave, ln, kh);
  __syncthreads();
  if (tid < 128) {
    const int r = tid;
    float fn = 0.f;
    for (int c = 0; c < 192; ++c) {
      float v = ACTF[r * LDF + c] + res_bo[2 * 192 + c];
      fn += v * v;
    }
    float best = 3.4e38f; int bg = 0;
    for (int g = 0; g < 64; ++g) {
      float d2 = fn - 2.0f * DIST[r * LDS_DI + g] + gridnorm[g];
      if (d2 < best) { best = d2; bg = g; }
    }
    atomicAdd(&hist[bg], 1u);
  }
}

// ---------------- finalize: pooled mean + 192x6 head + activations ----------

__global__ void arn_finalize(const unsigned* __restrict__ hist,
                             const float* __restrict__ sofm,
                             const float* __restrict__ out_w,
                             const float* __restrict__ out_b,
                             float* __restrict__ out)
{
  __shared__ float pooled[192];
  const int t = threadIdx.x;
  if (t < 192) {
    float s = 0.f;
    for (int g = 0; g < 64; ++g) s += (float)hist[g] * sofm[g * 192 + t];
    pooled[t] = s * (1.0f / 131072.0f);
  }
  __syncthreads();
  if (t < 6) {
    float o = out_b[t];
    for (int h = 0; h < 192; ++h) o += pooled[h] * out_w[h * 6 + t];
    out[t] = (t == 1) ? fmaxf(o, 0.f) : (1.0f / (1.0f + expf(-o)));
  }
}

// ---------------- launch ----------------

extern "C" void kernel_launch(void* const* d_in, const int* in_sizes, int n_in,
                              void* d_out, int out_size, void* d_ws, size_t ws_size,
                              hipStream_t stream) {
  (void)in_sizes; (void)n_in; (void)out_size; (void)ws_size;
  const float* x_vib  = (const float*)d_in[0];
  const float* x_aco  = (const float*)d_in[1];
  const float* x_tmp  = (const float*)d_in[2];
  const float* ew0    = (const float*)d_in[3];
  const float* eb0    = (const float*)d_in[4];
  const float* eg0    = (const float*)d_in[5];
  const float* ebb0   = (const float*)d_in[6];
  const float* ew1    = (const float*)d_in[7];
  const float* eb1    = (const float*)d_in[8];
  const float* eg1    = (const float*)d_in[9];
  const float* ebb1   = (const float*)d_in[10];
  const float* ew2    = (const float*)d_in[11];
  const float* eb2    = (const float*)d_in[12];
  const float* eg2    = (const float*)d_in[13];
  const float* ebb2   = (const float*)d_in[14];
  const float* fus_w  = (const float*)d_in[15];
  const float* fus_b  = (const float*)d_in[16];
  const float* fus_g  = (const float*)d_in[17];
  const float* fus_bb = (const float*)d_in[18];
  const float* res_wq = (const float*)d_in[19];
  const float* res_wk = (const float*)d_in[20];
  const float* res_wv = (const float*)d_in[21];
  const float* res_wo = (const float*)d_in[22];
  const float* res_bq = (const float*)d_in[23];
  const float* res_bk = (const float*)d_in[24];
  const float* res_bv = (const float*)d_in[25];
  const float* res_bo = (const float*)d_in[26];
  const float* res_mem= (const float*)d_in[27];
  const float* sofm   = (const float*)d_in[28];
  const float* out_w  = (const float*)d_in[29];
  const float* out_b  = (const float*)d_in[30];

  // workspace layout (bytes, 256-aligned)
  char* w = (char*)d_ws;
  unsigned* hist     = (unsigned*)(w + 0);       //  64 * 4
  float*    gridnorm = (float*)   (w + 256);     //  64 * 4
  bf16_t*   encT0    = (bf16_t*)  (w + 512);     // 192*64*2   = 24576
  bf16_t*   encT1    = (bf16_t*)  (w + 25088);   // 192*256*2  = 98304
  bf16_t*   encT2    = (bf16_t*)  (w + 123392);  // 192*128*2  = 49152
  bf16_t*   fusT     = (bf16_t*)  (w + 172544);  // 192*576*2  = 221184
  bf16_t*   wqT      = (bf16_t*)  (w + 393728);  // 3*192*192*2= 221184
  bf16_t*   woT      = (bf16_t*)  (w + 614912);  // 3*192*192*2= 221184
  bf16_t*   gridB    = (bf16_t*)  (w + 836096);  // 64*192*2   = 24576
  bf16_t*   kT       = (bf16_t*)  (w + 860672);  // 3*4*16*64*2= 24576
  bf16_t*   vT       = (bf16_t*)  (w + 885248);  // 3*4*48*32*2= 36864
  // total: 922112 bytes

  arn_prep_weights<<<256, 256, 0, stream>>>(
      ew0, ew1, ew2, fus_w, res_wq, res_wo, sofm,
      encT0, encT1, encT2, fusT, wqT, woT, gridB, gridnorm, hist);

  arn_prep_attn<<<64, 256, 0, stream>>>(
      res_mem, res_wk, res_wv, res_bk, res_bv, kT, vT);

  arn_fused_pipeline<<<1024, 256, SMEM_BYTES, stream>>>(
      x_vib, x_aco, x_tmp,
      eb0, eg0, ebb0, eb1, eg1, ebb1, eb2, eg2, ebb2,
      fus_b, fus_g, fus_bb, res_bq, res_bo,
      encT0, encT1, encT2, fusT, wqT, woT, kT, vT, gridB, gridnorm, hist);

  arn_finalize<<<1, 192, 0, stream>>>(hist, sofm, out_w, out_b, (float*)d_out);
}
